// ProtoNetHead_19387482375190
// MI455X (gfx1250) — matrix operand order repaired and checked by reading, not compiled, not run on hardware
//
#include <hip/hip_runtime.h>
#include <hip/hip_bf16.h>

// ---- CDNA5 (gfx1250) wave32 WMMA types ----
typedef __attribute__((ext_vector_type(16))) __bf16 v16bf;
typedef __attribute__((ext_vector_type(8)))  __bf16 v8bf;
typedef __attribute__((ext_vector_type(8)))  float  v8f;
typedef __attribute__((ext_vector_type(4)))  float  f32x4;

// Problem constants (from the reference setup)
#define DIN   1024
#define DPROJ 256
#define NCLS  64
#define NROWS (8 * 2048)   // B*Q = 16384
#define BROWS 32           // rows per block

// ---------------- prep kernels ----------------

// Transpose + convert proj_matrix [1024 x 256] f32 -> Wt [256 x 1024] bf16,
// so each WMMA B-fragment lane reads two runs of 8 contiguous K values.
__global__ void prep_wt(const float* __restrict__ W, __bf16* __restrict__ Wt) {
    int k = blockIdx.x;      // 0..1023
    int n = threadIdx.x;     // 0..255
    Wt[(size_t)n * DIN + k] = (__bf16)W[(size_t)k * DPROJ + n];
}

// mvec[n] = sum_k mean[k] * W[k][n]  (fold centering into the GEMM epilogue)
__global__ void prep_mvec(const float* __restrict__ W, const float* __restrict__ mean,
                          float* __restrict__ mvec) {
    int n = threadIdx.x;     // 0..255
    float acc = 0.f;
    for (int k = 0; k < DIN; ++k)
        acc = fmaf(mean[k], W[(size_t)k * DPROJ + n], acc);
    mvec[n] = acc;
}

// Convert prototypes to bf16 (already K-contiguous per class) and compute ||P_c||^2 in fp32.
__global__ void prep_proto(const float* __restrict__ P, __bf16* __restrict__ Pbf,
                           float* __restrict__ pnorm2) {
    int c = blockIdx.x;      // 0..63
    int d = threadIdx.x;     // 0..255
    float v = P[(size_t)c * DPROJ + d];
    Pbf[(size_t)c * DPROJ + d] = (__bf16)v;
    __shared__ float red[256];
    red[d] = v * v;
    __syncthreads();
    for (int s = 128; s > 0; s >>= 1) {
        if (d < s) red[d] += red[d + s];
        __syncthreads();
    }
    if (d == 0) pnorm2[c] = red[0];
}

// ---------------- fused main kernel ----------------
// Block: 256 threads (8 wave32). Covers 32 rows x 256 proj-cols.
// Stage 1: Z = X@Wt - mvec via v_wmma_f32_16x16x32_bf16 (fp32 accum, K=1024).
// Stage 2: normalize rows (cross-wave via LDS), then scores GEMM vs prototypes.
__launch_bounds__(256, 1)
__global__ void protonet_fused(const float* __restrict__ X,
                               const __bf16* __restrict__ Wt,
                               const __bf16* __restrict__ Pbf,
                               const float* __restrict__ mvec,
                               const float* __restrict__ pnorm2,
                               float* __restrict__ out) {
    __shared__ float  rowss[BROWS];
    __shared__ __bf16 Zb[BROWS][264];   // 264 halfs = 528 B row stride: 16B aligned, bank-spread

    const int tid  = threadIdx.x;
    const int wave = tid >> 5;
    const int lane = tid & 31;
    const int lo   = lane & 15;
    const int hi   = lane >> 4;          // 0 or 1 (K-half select in WMMA layouts)

    const int rowstrip = wave & 1;       // which 16-row strip (0..1)
    const int colgrp   = wave >> 1;      // which 64-col group (0..3)

    const int blockRow = blockIdx.x * BROWS;

    // ---- Stage 1: Z tile in accumulators ----
    // A fragment: lane holds row (blockRow + rowstrip*16 + lo), halfs = K {kb..kb+7, kb+16..kb+23}, kb = hi*8
    const size_t arow = (size_t)(blockRow + rowstrip * 16 + lo);
    const float* aptr = X + arow * DIN + hi * 8;

    // B fragments: lane holds column (colgrp*64 + t*16 + lo), halfs = K {kb..kb+7, kb+16..kb+23}, kb = hi*8
    const __bf16* bptr[4];
#pragma unroll
    for (int t = 0; t < 4; ++t)
        bptr[t] = Wt + (size_t)(colgrp * 64 + t * 16 + lo) * DIN + hi * 8;

    v8f acc[4] = {};

    for (int k = 0; k < DIN; k += 32) {
        // software prefetch next A chunk (speculative; dropped if OOB)
        __builtin_prefetch(aptr + k + 32, 0, 1);

        f32x4 a0 = *(const f32x4*)(aptr + k);
        f32x4 a1 = *(const f32x4*)(aptr + k + 4);
        f32x4 a2 = *(const f32x4*)(aptr + k + 16);
        f32x4 a3 = *(const f32x4*)(aptr + k + 20);
        v16bf af;
#pragma unroll
        for (int i = 0; i < 4; ++i) {
            af[i]      = (__bf16)a0[i];
            af[4 + i]  = (__bf16)a1[i];
            af[8 + i]  = (__bf16)a2[i];
            af[12 + i] = (__bf16)a3[i];
        }
#pragma unroll
        for (int t = 0; t < 4; ++t) {
            v16bf bf = __builtin_shufflevector(*(const v8bf*)(bptr[t] + k), *(const v8bf*)(bptr[t] + k + 16), 0, 1, 2, 3, 4, 5, 6, 7, 8, 9, 10, 11, 12, 13, 14, 15);
            acc[t] = __builtin_amdgcn_wmma_f32_16x16x32_bf16(
                false, af, false, bf, (short)0, acc[t], false, false);
        }
    }

    // epilogue: subtract mean projection (centering folded out of the GEMM)
    float mv[4];
#pragma unroll
    for (int t = 0; t < 4; ++t) mv[t] = mvec[colgrp * 64 + t * 16 + lo];
#pragma unroll
    for (int t = 0; t < 4; ++t)
#pragma unroll
        for (int r = 0; r < 8; ++r) acc[t][r] -= mv[t];

    // ---- cross-wave row sum-of-squares ----
    if (tid < BROWS) rowss[tid] = 0.f;
    __syncthreads();

    const int rbase = rowstrip * 16 + hi * 8;   // C/D layout: VGPR r -> row rbase + r, col = lo
#pragma unroll
    for (int r = 0; r < 8; ++r) {
        float ss = 0.f;
#pragma unroll
        for (int t = 0; t < 4; ++t) ss += acc[t][r] * acc[t][r];
        atomicAdd(&rowss[rbase + r], ss);       // ds_add_f32
    }
    __syncthreads();

    // normalize and stash bf16 Z-bar in LDS for stage-2 A fragments
#pragma unroll
    for (int r = 0; r < 8; ++r) {
        float inv = 1.0f / fmaxf(sqrtf(rowss[rbase + r]), 1e-12f);
#pragma unroll
        for (int t = 0; t < 4; ++t)
            Zb[rbase + r][colgrp * 64 + t * 16 + lo] = (__bf16)(acc[t][r] * inv);
    }
    __syncthreads();

    // ---- Stage 2: scores = -sqrt(1 + ||P||^2 - 2 * Zbar . P) ----
    const int rs2 = wave >> 2;                  // 16-row strip (0..1)
    const int ct2 = wave & 3;                   // 16-class tile (0..3)
    const int arow2 = rs2 * 16 + lo;
    const __bf16* pb = Pbf + (size_t)(ct2 * 16 + lo) * DPROJ + hi * 8;

    v8f acc2 = {};
#pragma unroll
    for (int kk = 0; kk < DPROJ; kk += 32) {
        v8bf z0 = *(const v8bf*)&Zb[arow2][kk + hi * 8];
        v8bf z1 = *(const v8bf*)&Zb[arow2][kk + hi * 8 + 16];
        v16bf af2 = __builtin_shufflevector(z0, z1,
            0, 1, 2, 3, 4, 5, 6, 7, 8, 9, 10, 11, 12, 13, 14, 15);
        v16bf bf2 = __builtin_shufflevector(*(const v8bf*)(pb + kk), *(const v8bf*)(pb + kk + 16), 0, 1, 2, 3, 4, 5, 6, 7, 8, 9, 10, 11, 12, 13, 14, 15);
        acc2 = __builtin_amdgcn_wmma_f32_16x16x32_bf16(
            false, af2, false, bf2, (short)0, acc2, false, false);
    }

    const int c = ct2 * 16 + lo;
    const float pn = pnorm2[c];
#pragma unroll
    for (int r = 0; r < 8; ++r) {
        int row = rs2 * 16 + hi * 8 + r;
        float d2 = fmaxf(1.0f + pn - 2.0f * acc2[r], 0.0f);
        out[(size_t)(blockRow + row) * NCLS + c] = -sqrtf(d2);
    }
}

// ---------------- launcher ----------------

extern "C" void kernel_launch(void* const* d_in, const int* in_sizes, int n_in,
                              void* d_out, int out_size, void* d_ws, size_t ws_size,
                              hipStream_t stream) {
    // setup_inputs order: X, prototypes, mean, proj_matrix (all fp32)
    const float* X     = (const float*)d_in[0];
    const float* proto = (const float*)d_in[1];
    const float* mean  = (const float*)d_in[2];
    const float* W     = (const float*)d_in[3];
    float* out = (float*)d_out;

    // workspace layout (all 16B-aligned offsets): ~547 KB total
    char* ws = (char*)d_ws;
    float*  mvec   = (float*)(ws);                              // 256 f32
    float*  pnorm2 = (float*)(ws + 1024);                       // 64 f32
    __bf16* Pbf    = (__bf16*)(ws + 2048);                      // 64*256 bf16 (32 KB)
    __bf16* Wt     = (__bf16*)(ws + 2048 + NCLS * DPROJ * 2);   // 256*1024 bf16 (512 KB)

    prep_wt   <<<DIN, DPROJ, 0, stream>>>(W, Wt);
    prep_mvec <<<1,   DPROJ, 0, stream>>>(W, mean, mvec);
    prep_proto<<<NCLS, DPROJ, 0, stream>>>(proto, Pbf, pnorm2);

    protonet_fused<<<NROWS / BROWS, 256, 0, stream>>>(X, Wt, Pbf, mvec, pnorm2, out);
}
